// GATEncoder_14551349199320
// MI455X (gfx1250) — matrix operand order repaired and checked by reading, not compiled, not run on hardware
//
#include <hip/hip_runtime.h>
#include <hip/hip_bf16.h>

// ---------------- problem constants ----------------
constexpr int N_NODES = 8192;
constexpr int IN_DIM  = 3000;
constexpr int HID     = 512;
constexpr int LAT     = 30;

#define AS1 __attribute__((address_space(1)))
#define AS3 __attribute__((address_space(3)))

// ---------------- WMMA / DMA types ----------------
typedef __attribute__((ext_vector_type(16))) __bf16 v16bf;
typedef __attribute__((ext_vector_type(8)))  float  v8f;
typedef __attribute__((ext_vector_type(4)))  int    v4i;   // b128 DMA chunk

union Frag16 {              // 16 bf16 = 8 packed dwords
    v16bf    v;
    unsigned u[8];
};

__device__ __forceinline__ unsigned short f32_to_bf16_u16(float f) {
    union { float f; unsigned u; } c; c.f = f;
    unsigned r = c.u + 0x7FFFu + ((c.u >> 16) & 1u);   // round-to-nearest-even
    return (unsigned short)(r >> 16);
}
__device__ __forceinline__ unsigned pack_bf16(float lo, float hi) {
    return (unsigned)f32_to_bf16_u16(lo) | ((unsigned)f32_to_bf16_u16(hi) << 16);
}
__device__ __forceinline__ float bf16_to_f32(unsigned short h) {
    union { unsigned u; float f; } c; c.u = ((unsigned)h) << 16;
    return c.f;
}

// =====================================================================
// Kernel 1: X = node_features @ W0   (bf16 WMMA, f32 accumulate)
// Output stored TRANSPOSED: XbfT[hid][node] so later stages can use raw
// async DMA (memory -> LDS) with K-contiguous fragment layout.
// grid = 512 blocks (16 rows each), 256 threads (8 waves x 64 cols)
// =====================================================================
__global__ __launch_bounds__(256) void k_gemm1(
    const float* __restrict__ Xin,          // [8192][3000]
    const float* __restrict__ W0,           // [3000][512]
    unsigned short* __restrict__ XbfT)      // out: [512][8192] bf16
{
    __shared__ unsigned Atile[16][17];      // 16x32 bf16, packed K-pairs
    __shared__ unsigned Bt[HID][17];        // 32x512 bf16, transposed packed K-pairs

    const int tid  = threadIdx.x;
    const int wave = tid >> 5;
    const int lane = tid & 31;
    const int rowBase = blockIdx.x * 16;

    const int m  = tid >> 4;                // 0..15 : A row this thread stages
    const int kp = tid & 15;                // K-pair index within tile

    v8f acc[4];
    for (int i = 0; i < 4; ++i)
        for (int r = 0; r < 8; ++r) acc[i][r] = 0.0f;

    const int numKT = (IN_DIM + 31) / 32;   // 94 (last tile zero-padded)
    for (int kt = 0; kt < numKT; ++kt) {
        const int k0 = kt * 32;
        // ---- stage A tile (16x32, f32 -> bf16) ----
        {
            const int kk = k0 + 2 * kp;
            float a0 = 0.f, a1 = 0.f;
            if (kk     < IN_DIM) a0 = Xin[(size_t)(rowBase + m) * IN_DIM + kk];
            if (kk + 1 < IN_DIM) a1 = Xin[(size_t)(rowBase + m) * IN_DIM + kk + 1];
            Atile[m][kp] = pack_bf16(a0, a1);
        }
        // ---- stage B tile (32x512, transposed-in-LDS K-pairs) ----
        {
            const int c0 = tid * 2;
            for (int r = 0; r < 32; r += 2) {
                float a0 = 0.f, a1 = 0.f, b0 = 0.f, b1 = 0.f;
                if (k0 + r     < IN_DIM) { a0 = W0[(size_t)(k0 + r) * HID + c0];
                                           a1 = W0[(size_t)(k0 + r) * HID + c0 + 1]; }
                if (k0 + r + 1 < IN_DIM) { b0 = W0[(size_t)(k0 + r + 1) * HID + c0];
                                           b1 = W0[(size_t)(k0 + r + 1) * HID + c0 + 1]; }
                Bt[c0    ][r >> 1] = pack_bf16(a0, b0);
                Bt[c0 + 1][r >> 1] = pack_bf16(a1, b1);
            }
        }
        __syncthreads();
        Frag16 af;
        {
            const int am = lane & 15;
            const int ab = (lane < 16) ? 0 : 4;      // K-pair base (K 0 / 8)
            for (int i = 0; i < 8; ++i) {
                const int p = (i < 4) ? (ab + i) : (8 + ab + (i - 4));
                af.u[i] = Atile[am][p];
            }
        }
        for (int nt = 0; nt < 4; ++nt) {
            Frag16 bf_;
            const int n  = wave * 64 + nt * 16 + (lane & 15);
            const int kb = (lane < 16) ? 0 : 4;      // K-pair base (K 0 / 8)
            for (int i = 0; i < 8; ++i) bf_.u[i] = Bt[n][kb + i + ((i < 4) ? 0 : 4)];
            acc[nt] = __builtin_amdgcn_wmma_f32_16x16x32_bf16(
                false, af.v, false, bf_.v, (short)0, acc[nt], false, false);
        }
        __syncthreads();
    }
    // ---- store X transposed as bf16 ----
    for (int nt = 0; nt < 4; ++nt)
        for (int r = 0; r < 8; ++r) {
            const int mm  = r + ((lane < 16) ? 0 : 8);
            const int col = wave * 64 + nt * 16 + (lane & 15);
            XbfT[(size_t)col * N_NODES + rowBase + mm] = f32_to_bf16_u16(acc[nt][r]);
        }
}

// =====================================================================
// Kernel 2: f0 = X@v0, f1 = X@v1 over transposed X (fully coalesced)
// =====================================================================
__global__ __launch_bounds__(256) void k_f01(
    const unsigned short* __restrict__ XbfT,   // [512][8192]
    const float* __restrict__ v0,
    const float* __restrict__ v1,
    float* __restrict__ f0,
    float* __restrict__ f1)
{
    const int i = blockIdx.x * 256 + threadIdx.x;     // node id
    float s0 = 0.f, s1 = 0.f;
    for (int k = 0; k < HID; ++k) {
        const float x = bf16_to_f32(XbfT[(size_t)k * N_NODES + i]);
        s0 += x * v0[k];
        s1 += x * v1[k];
    }
    f0[i] = s0;
    f1[i] = s1;
}

// =====================================================================
// Kernel 3: fused masked-softmax attention + aggregation + ELU
// 32 output rows per block (two A-tiles share every B-fragment ->
// 8 WMMAs per fragment gather), B-tile staged via async global->LDS DMA.
// grid = 256 blocks, 256 threads (8 waves x 64 cols)
// =====================================================================
__global__ __launch_bounds__(256) void k_attn(
    const float* __restrict__ graph,           // [8192][8192]
    const unsigned short* __restrict__ XbfT,   // [512][8192] bf16
    const float* __restrict__ f0,
    const float* __restrict__ f1,
    float* __restrict__ H)                     // out: [8192][512] f32 (post-ELU)
{
    __shared__ __align__(16) unsigned short BtT[HID][40]; // [col][K=32+pad]; row=80B (16B-aligned)
    __shared__ unsigned Atile[2][16][17];      // two 16x32 exp-weight tiles, packed pairs
    __shared__ float    rowsum[32];

    const int tid  = threadIdx.x;
    const int wave = tid >> 5;
    const int lane = tid & 31;
    const int rowBase = blockIdx.x * 32;

    const int m  = tid >> 4;                   // 0..15
    const int kp = tid & 15;
    const float f0m0 = f0[rowBase + m];
    const float f0m1 = f0[rowBase + 16 + m];

    float srow0 = 0.f, srow1 = 0.f;
    v8f acc[2][4];
    for (int a = 0; a < 2; ++a)
        for (int i = 0; i < 4; ++i)
            for (int r = 0; r < 8; ++r) acc[a][i][r] = 0.0f;

    for (int jt = 0; jt < N_NODES / 32; ++jt) {
        // ---- B tile: XbfT[0..511][jt*32 .. +32] -> LDS (32 KB) ----
        // 2048 x b128 chunks, 8 per thread, memory -> LDS with no VGPR round-trip.
#if __has_builtin(__builtin_amdgcn_global_load_async_to_lds_b128)
        for (int i = 0; i < 8; ++i) {
            const int c    = tid + i * 256;
            const int col  = c >> 2;
            const int part = c & 3;                      // 8 bf16 per part
            const unsigned short* g =
                XbfT + (size_t)col * N_NODES + jt * 32 + part * 8;
            __builtin_amdgcn_global_load_async_to_lds_b128(
                (AS1 v4i*)g,
                (AS3 v4i*)&BtT[col][part * 8], 0, 0);
        }
#else
        for (int i = 0; i < 8; ++i) {
            const int c    = tid + i * 256;
            const int col  = c >> 2;
            const int part = c & 3;
            const uint4* g = (const uint4*)(XbfT + (size_t)col * N_NODES
                                            + jt * 32 + part * 8);
            *(uint4*)&BtT[col][part * 8] = *g;
        }
#endif
        // ---- A tiles: exp-weights from graph, on the fly ----
        const int j0 = jt * 32 + 2 * kp;
        {
            const float g0 = graph[(size_t)(rowBase + m) * N_NODES + j0];
            const float g1 = graph[(size_t)(rowBase + m) * N_NODES + j0 + 1];
            float e0 = 0.f, e1 = 0.f;
            if (g0 != 0.f) { const float s = f0m0 + f1[j0];
                             e0 = __expf(1.f / (1.f + __expf(-s)) - 0.5f); }
            if (g1 != 0.f) { const float s = f0m0 + f1[j0 + 1];
                             e1 = __expf(1.f / (1.f + __expf(-s)) - 0.5f); }
            srow0 += e0 + e1;
            Atile[0][m][kp] = pack_bf16(e0, e1);
        }
        {
            const float g0 = graph[(size_t)(rowBase + 16 + m) * N_NODES + j0];
            const float g1 = graph[(size_t)(rowBase + 16 + m) * N_NODES + j0 + 1];
            float e0 = 0.f, e1 = 0.f;
            if (g0 != 0.f) { const float s = f0m1 + f1[j0];
                             e0 = __expf(1.f / (1.f + __expf(-s)) - 0.5f); }
            if (g1 != 0.f) { const float s = f0m1 + f1[j0 + 1];
                             e1 = __expf(1.f / (1.f + __expf(-s)) - 0.5f); }
            srow1 += e0 + e1;
            Atile[1][m][kp] = pack_bf16(e0, e1);
        }
        // keep the 256 MB graph stream ahead of compute
        if (jt + 1 < N_NODES / 32)
            __builtin_prefetch(&graph[(size_t)(rowBase + m) * N_NODES + j0 + 32], 0, 1);

#if __has_builtin(__builtin_amdgcn_global_load_async_to_lds_b128)
#if __has_builtin(__builtin_amdgcn_s_wait_asynccnt)
        __builtin_amdgcn_s_wait_asynccnt(0);
#else
        asm volatile("s_wait_asynccnt 0x0" ::: "memory");
#endif
#endif
        __syncthreads();

        // ---- fragments ----
        Frag16 a0f, a1f;
        {
            const int am = lane & 15;
            const int ab = (lane < 16) ? 0 : 4;
            for (int i = 0; i < 8; ++i) {
                const int p = (i < 4) ? (ab + i) : (8 + ab + (i - 4));
                a0f.u[i] = Atile[0][am][p];
                a1f.u[i] = Atile[1][am][p];
            }
        }
        for (int nt = 0; nt < 4; ++nt) {
            Frag16 bf_;
            const int n  = wave * 64 + nt * 16 + (lane & 15);
            const int kb = (lane < 16) ? 0 : 8;        // K base
            for (int i = 0; i < 8; ++i)
                bf_.u[i] = *(const unsigned*)&BtT[n][kb + 2 * i + ((i < 4) ? 0 : 8)];
            acc[0][nt] = __builtin_amdgcn_wmma_f32_16x16x32_bf16(
                false, a0f.v, false, bf_.v, (short)0, acc[0][nt], false, false);
            acc[1][nt] = __builtin_amdgcn_wmma_f32_16x16x32_bf16(
                false, a1f.v, false, bf_.v, (short)0, acc[1][nt], false, false);
        }
        __syncthreads();
    }

    // ---- reduce per-row exp sums (thread t owns row t/16 of each tile) ----
    for (int msk = 8; msk >= 1; msk >>= 1) {
        srow0 += __shfl_xor(srow0, msk, 16);
        srow1 += __shfl_xor(srow1, msk, 16);
    }
    if ((lane & 15) == 0) {
        rowsum[wave * 2 + (lane >> 4)]      = srow0;
        rowsum[16 + wave * 2 + (lane >> 4)] = srow1;
    }
    __syncthreads();

    // ---- normalize + ELU + store ----
    for (int at = 0; at < 2; ++at)
        for (int nt = 0; nt < 4; ++nt)
            for (int r = 0; r < 8; ++r) {
                const int mm = at * 16 + r + ((lane < 16) ? 0 : 8);
                float v = acc[at][nt][r] / fmaxf(rowsum[mm], 1e-30f);
                v = (v > 0.f) ? v : (__expf(v) - 1.f);
                H[(size_t)(rowBase + mm) * HID + wave * 64 + nt * 16 + (lane & 15)] = v;
            }
}

// =====================================================================
// Kernel 4: out = H @ W1   (tiny N=30; 0.25 GFLOP -> plain VALU)
// =====================================================================
__global__ __launch_bounds__(256) void k_final(
    const float* __restrict__ H,
    const float* __restrict__ W1,           // [512][30]
    float* __restrict__ out)                // [8192][30]
{
    const int idx = blockIdx.x * 256 + threadIdx.x;
    if (idx >= N_NODES * LAT) return;
    const int i = idx / LAT, c = idx % LAT;
    const float* h = H + (size_t)i * HID;
    float s = 0.f;
    for (int k = 0; k < HID; ++k) s += h[k] * W1[k * LAT + c];
    out[idx] = s;
}

// =====================================================================
extern "C" void kernel_launch(void* const* d_in, const int* in_sizes, int n_in,
                              void* d_out, int out_size, void* d_ws, size_t ws_size,
                              hipStream_t stream) {
    (void)in_sizes; (void)n_in; (void)out_size; (void)ws_size;
    const float* Xin   = (const float*)d_in[0];   // node_features [8192*3000]
    const float* graph = (const float*)d_in[1];   // [8192*8192]
    const float* W0    = (const float*)d_in[2];   // [3000*512]
    const float* v0    = (const float*)d_in[3];   // [512]
    const float* v1    = (const float*)d_in[4];   // [512]
    const float* W1    = (const float*)d_in[5];   // [512*30]
    float* out = (float*)d_out;

    char* ws = (char*)d_ws;
    unsigned short* XbfT = (unsigned short*)(ws);                     // 8 MB  [512][8192]
    float* H  = (float*)(ws + (size_t)N_NODES * HID * 2);             // 16 MB [8192][512]
    float* f0 = (float*)(ws + (size_t)N_NODES * HID * 2
                            + (size_t)N_NODES * HID * 4);             // 32 KB
    float* f1 = f0 + N_NODES;                                         // 32 KB

    k_gemm1<<<N_NODES / 16, 256, 0, stream>>>(Xin, W0, XbfT);
    k_f01  <<<N_NODES / 256, 256, 0, stream>>>(XbfT, v0, v1, f0, f1);
    k_attn <<<N_NODES / 32, 256, 0, stream>>>(graph, XbfT, f0, f1, H);
    k_final<<<(N_NODES * LAT + 255) / 256, 256, 0, stream>>>(H, W1, out);
}
